// CausalSelfAttention_12438225290011
// MI455X (gfx1250) — matrix operand order repaired and checked by reading, not compiled, not run on hardware
//
#include <hip/hip_runtime.h>
#include <cstdint>
#include <cstddef>

#define SEQ    2048
#define DIM    2048
#define NHEAD  16
#define HD     128

typedef __attribute__((ext_vector_type(16))) __bf16 bf16x16;
typedef __attribute__((ext_vector_type(8)))  float  f32x8;

union Frag16 {
  uint4   u[2];
  bf16x16 v;
};

__device__ __forceinline__ __bf16 f2bf(float f) {
  union { float f; unsigned u; } x; x.f = f;
  unsigned r = x.u + 0x7FFFu + ((x.u >> 16) & 1u);   // round-to-nearest-even
  unsigned short h = (unsigned short)(r >> 16);
  union { unsigned short s; __bf16 b; } y; y.s = h;
  return y.b;
}

__device__ __forceinline__ uint32_t lds_addr(const void* p) {
  // flat shared pointer: addr[31:0] is the LDS byte address (ISA 10.2 aperture rules)
  return (uint32_t)(uintptr_t)p;
}

__device__ __forceinline__ void async_load_b128(uint32_t lds, const void* gptr) {
  asm volatile("global_load_async_to_lds_b128 %0, %1, off"
               :: "v"(lds), "v"(gptr)
               : "memory");
}

__device__ __forceinline__ void wait_async0() {
  asm volatile("s_wait_asynccnt 0x0" ::: "memory");
}

// ---------------------------------------------------------------------------
// Elementwise f32 -> bf16 convert
// ---------------------------------------------------------------------------
__global__ __launch_bounds__(256)
void cvt_f32_bf16(const float* __restrict__ s, __bf16* __restrict__ d, int n) {
  int i = blockIdx.x * 256 + threadIdx.x;
  if (i < n) d[i] = f2bf(s[i]);
}

// ---------------------------------------------------------------------------
// Transpose f32[R][C] -> bf16[C][R]
// ---------------------------------------------------------------------------
__global__ __launch_bounds__(256)
void transpose_cvt(const float* __restrict__ src, __bf16* __restrict__ dst,
                   int R, int C) {
  __shared__ float tile[32][33];
  const int tx = threadIdx.x, ty = threadIdx.y;
  const int x  = blockIdx.x * 32 + tx;
  const int y0 = blockIdx.y * 32;
  for (int i = ty; i < 32; i += 8)
    tile[i][tx] = src[(size_t)(y0 + i) * C + x];
  __syncthreads();
  const int ox = blockIdx.y * 32 + tx;
  for (int i = ty; i < 32; i += 8)
    dst[(size_t)(blockIdx.x * 32 + i) * R + ox] = f2bf(tile[tx][i]);
}

// ---------------------------------------------------------------------------
// C[M][N] = A[M][K] @ Bt[N][K]^T + bias[N]      (bf16 in, f32 out, WMMA)
// block: 256 threads = 8 waves (4 in M x 2 in N), tile 128x128, K step 32.
// Double-buffered LDS staging through GLOBAL_LOAD_ASYNC_TO_LDS_B128.
// ---------------------------------------------------------------------------
__global__ __launch_bounds__(256)
void gemm_bf16_wmma(const __bf16* __restrict__ A, const __bf16* __restrict__ Bt,
                    const float* __restrict__ bias, float* __restrict__ C,
                    int M, int N, int K) {
  // Sm[buf][0] = A tile, Sm[buf][1] = B tile; row stride 40 halves (80B, bank-spread)
  __shared__ __align__(16) __bf16 Sm[2][2][128 * 40];
  const int tid  = threadIdx.x;
  const int w    = tid >> 5;
  const int lane = tid & 31;
  const int ln   = lane & 15;
  const int hh   = lane >> 4;
  const int wm   = w >> 1;              // 0..3
  const int wn   = w & 1;               // 0..1
  const int m0   = blockIdx.y * 128;
  const int n0   = blockIdx.x * 128;

  // each thread owns two 16B chunks per tile: rows row0 and row0+64
  const int row0 = tid >> 2;            // 0..63
  const int off  = (tid & 3) * 8;       // bf16 element offset 0,8,16,24
  const __bf16* pA0 = A  + (size_t)(m0 + row0) * K + off;
  const __bf16* pA1 = pA0 + (size_t)64 * K;
  const __bf16* pB0 = Bt + (size_t)(n0 + row0) * K + off;
  const __bf16* pB1 = pB0 + (size_t)64 * K;
  const uint32_t lA0[2] = { lds_addr(&Sm[0][0][row0 * 40 + off]),
                            lds_addr(&Sm[1][0][row0 * 40 + off]) };
  const uint32_t lB0[2] = { lds_addr(&Sm[0][1][row0 * 40 + off]),
                            lds_addr(&Sm[1][1][row0 * 40 + off]) };
  const uint32_t rstep = 64u * 40u * 2u;   // 64 rows in bytes

  f32x8 acc[2][4];
  for (int i = 0; i < 2; ++i)
    for (int j = 0; j < 4; ++j)
      for (int r = 0; r < 8; ++r) acc[i][j][r] = 0.f;

  // prologue: stage k0 = 0 into buffer 0
  async_load_b128(lA0[0],         pA0);
  async_load_b128(lA0[0] + rstep, pA1);
  async_load_b128(lB0[0],         pB0);
  async_load_b128(lB0[0] + rstep, pB1);

  int buf = 0;
  for (int k0 = 0; k0 < K; k0 += 32, buf ^= 1) {
    wait_async0();          // my async stores to Sm[buf] have landed
    __syncthreads();        // everyone's have landed; prev reads of Sm[buf] done
    if (k0 + 32 < K) {
      const int nb = buf ^ 1;
      async_load_b128(lA0[nb],         pA0 + k0 + 32);
      async_load_b128(lA0[nb] + rstep, pA1 + k0 + 32);
      async_load_b128(lB0[nb],         pB0 + k0 + 32);
      async_load_b128(lB0[nb] + rstep, pB1 + k0 + 32);
    }

    const __bf16* As = &Sm[buf][0][0];
    const __bf16* Bs = &Sm[buf][1][0];
    Frag16 af[2], bfr[4];
    for (int i = 0; i < 2; ++i) {
      const int row = wm * 32 + i * 16 + ln;
      af[i].u[0] = *(const uint4*)&As[row * 40 + 8 * hh];
      af[i].u[1] = *(const uint4*)&As[row * 40 + 16 + 8 * hh];
    }
    for (int j = 0; j < 4; ++j) {
      const int row = wn * 64 + j * 16 + ln;
      bfr[j].u[0] = *(const uint4*)&Bs[row * 40 + 8 * hh];
      bfr[j].u[1] = *(const uint4*)&Bs[row * 40 + 16 + 8 * hh];
    }
    for (int i = 0; i < 2; ++i)
      for (int j = 0; j < 4; ++j)
        acc[i][j] = __builtin_amdgcn_wmma_f32_16x16x32_bf16(
            false, af[i].v, false, bfr[j].v, (short)0, acc[i][j], false, false);
  }

  for (int i = 0; i < 2; ++i) {
    const int rbase = m0 + wm * 32 + i * 16 + 8 * hh;
    for (int j = 0; j < 4; ++j) {
      const int n  = n0 + wn * 64 + j * 16 + ln;
      const float bv = bias[n];
      for (int r = 0; r < 8; ++r)
        C[(size_t)(rbase + r) * N + n] = acc[i][j][r] + bv;
    }
  }
}

// ---------------------------------------------------------------------------
// RMSNorm (full 2048-row) + RoPE, f32 in -> bf16 out. One block per row.
// ---------------------------------------------------------------------------
__global__ __launch_bounds__(256)
void normrope(const float* __restrict__ src, const float* __restrict__ g,
              const float* __restrict__ cosp, const float* __restrict__ sinp,
              __bf16* __restrict__ dst) {
  __shared__ float red[256];
  const int t   = blockIdx.x;
  const int tid = threadIdx.x;
  const float* row = src + (size_t)t * DIM;
  float v[8];
  float ss = 0.f;
  for (int e = 0; e < 8; ++e) { v[e] = row[tid * 8 + e]; ss += v[e] * v[e]; }
  red[tid] = ss;
  __syncthreads();
  for (int st = 128; st > 0; st >>= 1) {
    if (tid < st) red[tid] += red[tid + st];
    __syncthreads();
  }
  const float rms = rsqrtf(red[0] / (float)DIM + 1e-6f);
  for (int e = 0; e < 8; e += 2) {
    const int i = tid * 8 + e;
    const int p = (i & (HD - 1)) >> 1;           // rotary pair index in head
    const float c  = cosp[(size_t)t * (HD / 2) + p];
    const float sn = sinp[(size_t)t * (HD / 2) + p];
    const float a = v[e] * rms * g[i];
    const float b = v[e + 1] * rms * g[i + 1];
    dst[(size_t)t * DIM + i]     = f2bf(a * c - b * sn);
    dst[(size_t)t * DIM + i + 1] = f2bf(a * sn + b * c);
  }
}

// ---------------------------------------------------------------------------
// Causal flash attention. Q,K: bf16[T][DIM] (row-major per head);
// Vt: bf16[DIM][T] (transposed). O: bf16[T][DIM].
// grid (T/64, NHEAD), block 128 = 4 waves, each wave owns 16 q-rows.
// ---------------------------------------------------------------------------
__global__ __launch_bounds__(128)
void attn_flash_wmma(const __bf16* __restrict__ Q, const __bf16* __restrict__ Kc,
                     const __bf16* __restrict__ Vt, __bf16* __restrict__ O) {
  __shared__ __align__(16) __bf16 Plds[4 * 16 * 40];
  const int tid  = threadIdx.x;
  const int w    = tid >> 5;
  const int lane = tid & 31;
  const int ln   = lane & 15;
  const int hh   = lane >> 4;
  const int h    = blockIdx.y;
  const int q0w  = blockIdx.x * 64 + w * 16;
  const float SCALE = 0.08838834764831845f;       // 1/sqrt(128)

  Frag16 qf[4];
  {
    const __bf16* qr = Q + (size_t)(q0w + ln) * DIM + h * HD;
    for (int c = 0; c < 4; ++c) {
      qf[c].u[0] = *(const uint4*)&qr[32 * c + 16 * hh];
      qf[c].u[1] = *(const uint4*)&qr[32 * c + 16 * hh + 8];
    }
  }

  float m[8], s[8];
  f32x8 o[8];
  for (int r = 0; r < 8; ++r) { m[r] = -1e30f; s[r] = 0.f; }
  for (int nd = 0; nd < 8; ++nd)
    for (int r = 0; r < 8; ++r) o[nd][r] = 0.f;

  // running pointers (no per-iteration 64-bit multiplies)
  const __bf16* kp0 = Kc + (size_t)ln * DIM + h * HD + 16 * hh;        // j = 0
  const __bf16* kp1 = kp0 + (size_t)16 * DIM;                          // j = 1
  const __bf16* vp  = Vt + (size_t)(h * HD + ln) * SEQ + 16 * hh;      // nd = 0
  const size_t  kstep = (size_t)32 * DIM;

  __bf16* P = Plds + w * (16 * 40);
  const int nkb = blockIdx.x * 2 + 2;             // cover keys [0, qtile_end]
  for (int kbi = 0; kbi < nkb; ++kbi) {
    const int kb = kbi * 32;
    f32x8 sc[2];
    for (int j = 0; j < 2; ++j)
      for (int r = 0; r < 8; ++r) sc[j][r] = 0.f;

    // scores = Q @ K^T
    for (int c = 0; c < 4; ++c) {
      Frag16 kf;
      kf.u[0] = *(const uint4*)&kp0[32 * c];
      kf.u[1] = *(const uint4*)&kp0[32 * c + 8];
      sc[0] = __builtin_amdgcn_wmma_f32_16x16x32_bf16(
          false, qf[c].v, false, kf.v, (short)0, sc[0], false, false);
    }
    for (int c = 0; c < 4; ++c) {
      Frag16 kf;
      kf.u[0] = *(const uint4*)&kp1[32 * c];
      kf.u[1] = *(const uint4*)&kp1[32 * c + 8];
      sc[1] = __builtin_amdgcn_wmma_f32_16x16x32_bf16(
          false, qf[c].v, false, kf.v, (short)0, sc[1], false, false);
    }
    kp0 += kstep;
    kp1 += kstep;

    // scale + causal mask
    for (int j = 0; j < 2; ++j) {
      const int ka = kb + j * 16 + ln;
      for (int r = 0; r < 8; ++r) {
        const int qa = q0w + r + 8 * hh;
        const float val = sc[j][r] * SCALE;
        sc[j][r] = (ka <= qa) ? val : -1e30f;
      }
    }

    // online softmax update (row reductions across the 16-lane half)
    float pn[2][8];
    for (int r = 0; r < 8; ++r) {
      float t = fmaxf(sc[0][r], sc[1][r]);
      t = fmaxf(t, __shfl_xor(t, 1, 16));
      t = fmaxf(t, __shfl_xor(t, 2, 16));
      t = fmaxf(t, __shfl_xor(t, 4, 16));
      t = fmaxf(t, __shfl_xor(t, 8, 16));
      const float mn = fmaxf(m[r], t);
      const float al = __expf(m[r] - mn);
      const float p0 = __expf(sc[0][r] - mn);
      const float p1 = __expf(sc[1][r] - mn);
      float rs = p0 + p1;
      rs += __shfl_xor(rs, 1, 16);
      rs += __shfl_xor(rs, 2, 16);
      rs += __shfl_xor(rs, 4, 16);
      rs += __shfl_xor(rs, 8, 16);
      s[r] = s[r] * al + rs;
      m[r] = mn;
      pn[0][r] = p0; pn[1][r] = p1;
      for (int nd = 0; nd < 8; ++nd) o[nd][r] *= al;
    }

    // C-layout P -> LDS -> A-layout fragment
    for (int j = 0; j < 2; ++j)
      for (int r = 0; r < 8; ++r)
        P[(r + 8 * hh) * 40 + j * 16 + ln] = f2bf(pn[j][r]);
    __syncthreads();
    Frag16 pf;
    pf.u[0] = *(const uint4*)&P[ln * 40 + 16 * hh];
    pf.u[1] = *(const uint4*)&P[ln * 40 + 16 * hh + 8];

    // O += P @ V
    for (int nd = 0; nd < 8; ++nd) {
      Frag16 vfr;
      const __bf16* vr = vp + (size_t)(nd * 16) * SEQ + kb;
      vfr.u[0] = *(const uint4*)&vr[0];
      vfr.u[1] = *(const uint4*)&vr[8];
      o[nd] = __builtin_amdgcn_wmma_f32_16x16x32_bf16(
          false, pf.v, false, vfr.v, (short)0, o[nd], false, false);
    }
    __syncthreads();
  }

  __bf16* ob = O + (size_t)(q0w + 8 * hh) * DIM + h * HD + ln;
  for (int nd = 0; nd < 8; ++nd)
    for (int r = 0; r < 8; ++r)
      ob[(size_t)r * DIM + nd * 16] = f2bf(o[nd][r] * (1.f / s[r]));
}

// ---------------------------------------------------------------------------
extern "C" void kernel_launch(void* const* d_in, const int* in_sizes, int n_in,
                              void* d_out, int out_size, void* d_ws, size_t ws_size,
                              hipStream_t stream) {
  const float* x  = (const float*)d_in[0];
  const float* fc = (const float*)d_in[1];
  const float* fs = (const float*)d_in[2];
  const float* Wq = (const float*)d_in[3];
  const float* bq = (const float*)d_in[4];
  const float* Wk = (const float*)d_in[5];
  const float* bk = (const float*)d_in[6];
  const float* Wv = (const float*)d_in[7];
  const float* bv = (const float*)d_in[8];
  const float* Wo = (const float*)d_in[9];
  const float* bo = (const float*)d_in[10];
  const float* gq = (const float*)d_in[11];
  const float* gk = (const float*)d_in[12];
  float* out = (float*)d_out;
  (void)in_sizes; (void)n_in; (void)out_size; (void)ws_size;

  const size_t NE = (size_t)SEQ * DIM;
  char* p = (char*)d_ws;
  auto take = [&](size_t bytes) {
    char* r = p;
    p += (bytes + 255) & ~(size_t)255;
    return r;
  };
  __bf16* xb   = (__bf16*)take(NE * 2);
  __bf16* WqT  = (__bf16*)take(NE * 2);
  __bf16* WkT  = (__bf16*)take(NE * 2);
  __bf16* WvT  = (__bf16*)take(NE * 2);
  __bf16* WoT  = (__bf16*)take(NE * 2);
  float*  qf   = (float*) take(NE * 4);
  float*  kf   = (float*) take(NE * 4);
  float*  vf   = (float*) take(NE * 4);
  __bf16* qb   = (__bf16*)take(NE * 2);
  __bf16* kb2  = (__bf16*)take(NE * 2);
  __bf16* vTb  = (__bf16*)take(NE * 2);
  __bf16* attb = (__bf16*)take(NE * 2);

  // 1. precision prep
  cvt_f32_bf16<<<dim3((unsigned)(NE / 256)), 256, 0, stream>>>(x, xb, (int)NE);
  dim3 tb(32, 8), tg(DIM / 32, DIM / 32);
  transpose_cvt<<<tg, tb, 0, stream>>>(Wq, WqT, DIM, DIM);
  transpose_cvt<<<tg, tb, 0, stream>>>(Wk, WkT, DIM, DIM);
  transpose_cvt<<<tg, tb, 0, stream>>>(Wv, WvT, DIM, DIM);
  transpose_cvt<<<tg, tb, 0, stream>>>(Wo, WoT, DIM, DIM);

  // 2. q/k/v projections
  dim3 gg(DIM / 128, SEQ / 128);
  gemm_bf16_wmma<<<gg, 256, 0, stream>>>(xb, WqT, bq, qf, SEQ, DIM, DIM);
  gemm_bf16_wmma<<<gg, 256, 0, stream>>>(xb, WkT, bk, kf, SEQ, DIM, DIM);
  gemm_bf16_wmma<<<gg, 256, 0, stream>>>(xb, WvT, bv, vf, SEQ, DIM, DIM);

  // 3. rmsnorm + rope for q/k, transpose v
  normrope<<<SEQ, 256, 0, stream>>>(qf, gq, fc, fs, qb);
  normrope<<<SEQ, 256, 0, stream>>>(kf, gk, fc, fs, kb2);
  transpose_cvt<<<tg, tb, 0, stream>>>(vf, vTb, SEQ, DIM);

  // 4. causal flash attention
  attn_flash_wmma<<<dim3(SEQ / 64, NHEAD), 128, 0, stream>>>(qb, kb2, vTb, attb);

  // 5. output projection
  gemm_bf16_wmma<<<gg, 256, 0, stream>>>(attb, WoT, bo, out, SEQ, DIM, DIM);
}
